// GatingGCN_52673478918494
// MI455X (gfx1250) — matrix operand order repaired and checked by reading, not compiled, not run on hardware
//
#include <hip/hip_runtime.h>
#include <hip/hip_bf16.h>

typedef __attribute__((ext_vector_type(16))) __bf16 v16bf;
typedef __attribute__((ext_vector_type(8)))  float  v8f;

#define HID 128
#define NEXP 8
#define NGRAPH 64

// ---------------- utility kernels ----------------
__global__ void k_zero(float* p, long n) {
  long i = (long)blockIdx.x * blockDim.x + threadIdx.x;
  if (i < n) p[i] = 0.0f;
}

// Pre-permute W2 (f32 [K=128, N=128] row-major) into bf16 fragment order:
// wp[ct][kbi][lane][e], ct = N/16 tile, kbi = K/32 step,
// lane: half=lane>>4 selects K 0-15 vs 16-31 within the 32-step, n = ct*16 + (lane&15).
__global__ void k_permute_w(const float* W, __bf16* wp) {
  int i = blockIdx.x * blockDim.x + threadIdx.x;
  if (i >= HID * HID) return;
  int e    = i & 15;
  int lane = (i >> 4) & 31;
  int kbi  = (i >> 9) & 3;
  int ct   = i >> 11;
  int k = kbi * 32 + (lane >> 4) * 16 + e;
  int n = ct * 16 + (lane & 15);
  wp[i] = (__bf16)W[k * HID + n];
}

// degree of each destination node (self-loop added later as +1)
__global__ void k_deg(const long long* dst, float* deg, int E) {
  int e = blockIdx.x * blockDim.x + threadIdx.x;
  if (e < E) atomicAdd(&deg[dst[e]], 1.0f);
}

__global__ void k_dis(const float* deg, float* dis, int N) {
  int i = blockIdx.x * blockDim.x + threadIdx.x;
  if (i < N) dis[i] = rsqrtf(deg[i] + 1.0f);   // +1 self-loop; always > 0
}

// h1 = [an | pos] @ W1   (K=4, pure VALU; 0.1 GFLOP, negligible)
__global__ void k_fc1(const float* an, const float* pos, const float* W1,
                      float* h1, int N) {
  int t = blockIdx.x * blockDim.x + threadIdx.x;
  if (t >= N * HID) return;
  int n = t >> 7, f = t & 127;
  float acc = an[n]        * W1[f]
            + pos[3*n + 0] * W1[128 + f]
            + pos[3*n + 1] * W1[256 + f]
            + pos[3*n + 2] * W1[384 + f];
  h1[t] = acc;
}

// scatter: out[dst] += h[src] * dis[src]*dis[dst]   (one thread = 16 features)
__global__ void k_agg(const long long* src, const long long* dst, const float* dis,
                      const float* h, float* out, int E) {
  int t = blockIdx.x * blockDim.x + threadIdx.x;
  if (t >= E * 8) return;
  int e = t >> 3, g = t & 7;
  long s = (long)src[e], d = (long)dst[e];
  float norm = dis[s] * dis[d];
  const float4* hp = (const float4*)(h + s * HID + g * 16);
  float* op = out + d * HID + g * 16;
#pragma unroll
  for (int q = 0; q < 4; ++q) {
    float4 v = hp[q];
    atomicAdd(op + q * 4 + 0, v.x * norm);
    atomicAdd(op + q * 4 + 1, v.y * norm);
    atomicAdd(op + q * 4 + 2, v.z * norm);
    atomicAdd(op + q * 4 + 3, v.w * norm);
  }
}

// epilogue: add self-loop message (h * dis^2), bias, optional ReLU (in place)
__global__ void k_self_bias(const float* h, const float* dis, const float* b,
                            float* x, int N, int do_relu) {
  int t = blockIdx.x * blockDim.x + threadIdx.x;
  if (t >= N * HID) return;
  int n = t >> 7, f = t & 127;
  float di = dis[n];
  float v = x[t] + h[t] * di * di + b[f];
  x[t] = do_relu ? fmaxf(v, 0.0f) : v;
}

// ---------------- WMMA GEMM: Y[N,128] = X[N,128] @ W[128,128]
// 256 threads = 8 wave32; block tile = 64 rows x 128 cols; wave = one 16-col strip,
// 4 M-tiles per wave -> 16 v_wmma per wave. B fragments come pre-permuted from
// global (L2-hot, contiguous 32B per lane); A staged once in LDS (16 KB).
__global__ void __launch_bounds__(256) k_gemm_wmma(const float* X, const __bf16* Wp,
                                                   float* Y, int N) {
  __shared__ __bf16 Xl[64 * HID];   // 16 KB of 320 KB WGP LDS
  const int tid  = threadIdx.x;
  const int row0 = blockIdx.x * 64;

  for (int i = tid; i < 64 * HID; i += 256) {
    int r = row0 + (i >> 7);
    Xl[i] = (r < N) ? (__bf16)X[(long)r * HID + (i & 127)] : (__bf16)0.0f;
  }
  __syncthreads();

  const int lane = tid & 31;
  const int wave = tid >> 5;          // col tile ct = wave
  const int half = lane >> 4;
  const int nloc = lane & 15;
  const int col  = (wave << 4) + nloc;

  v8f c[4] = {v8f{}, v8f{}, v8f{}, v8f{}};

#pragma unroll
  for (int kbi = 0; kbi < 4; ++kbi) {
    const int kb = kbi * 32;
    // B fragment: one contiguous 32B per-lane load from pre-permuted global W
    v16bf b = *(const v16bf*)(Wp + (((wave * 4 + kbi) * 32 + lane) << 4));
#pragma unroll
    for (int mt = 0; mt < 4; ++mt) {
      const int m = (mt << 4) + nloc;
      // A 16x32 bf16 layout: two contiguous 16B chunks per lane -> ds_load_b128 x2
      const __bf16* ap = Xl + m * HID + kb + 16 * half;
      v16bf a;
#pragma unroll
      for (int e = 0; e < 8; ++e) { a[e] = ap[e]; a[e + 8] = ap[8 + e]; }
      c[mt] = __builtin_amdgcn_wmma_f32_16x16x32_bf16(false, a, false, b,
                                                      (short)0, c[mt], false, false);
    }
  }

  // C/D 16x16 f32 layout: VGPR r -> M = r + 8*half, N = lane%16
#pragma unroll
  for (int mt = 0; mt < 4; ++mt) {
#pragma unroll
    for (int r = 0; r < 8; ++r) {
      int m = row0 + (mt << 4) + r + 8 * half;
      if (m < N) Y[(long)m * HID + col] = c[mt][r];
    }
  }
}

// logits = X @ Wlin + blin fused with segment-sum pooling (1 thread = 1 node, 8 experts)
__global__ void k_lin_pool(const float* X, const float* Wlin, const float* blin,
                           const long long* batch, float* psum, float* pcnt, int N) {
  int n = blockIdx.x * blockDim.x + threadIdx.x;
  if (n >= N) return;
  const float* xr = X + (long)n * HID;
  float acc[NEXP];
#pragma unroll
  for (int e = 0; e < NEXP; ++e) acc[e] = blin[e];
  for (int k = 0; k < HID; ++k) {
    float xv = xr[k];
#pragma unroll
    for (int e = 0; e < NEXP; ++e) acc[e] += xv * Wlin[k * NEXP + e];
  }
  long g = (long)batch[n];
#pragma unroll
  for (int e = 0; e < NEXP; ++e) atomicAdd(&psum[g * NEXP + e], acc[e]);
  atomicAdd(&pcnt[g], 1.0f);
}

// mean + log_softmax per graph (one thread per graph row)
__global__ void k_finalize(const float* psum, const float* pcnt, float* out) {
  int g = threadIdx.x;
  if (g >= NGRAPH) return;
  float cnt = fmaxf(pcnt[g], 1.0f);
  float v[NEXP];
  float mx = -3.0e38f;
  for (int e = 0; e < NEXP; ++e) { v[e] = psum[g * NEXP + e] / cnt; mx = fmaxf(mx, v[e]); }
  float s = 0.0f;
  for (int e = 0; e < NEXP; ++e) s += expf(v[e] - mx);
  float lse = logf(s);
  for (int e = 0; e < NEXP; ++e) out[g * NEXP + e] = v[e] - mx - lse;
}

// ---------------- launch ----------------
extern "C" void kernel_launch(void* const* d_in, const int* in_sizes, int n_in,
                              void* d_out, int out_size, void* d_ws, size_t ws_size,
                              hipStream_t stream) {
  const float*     an    = (const float*)d_in[0];
  const float*     pos   = (const float*)d_in[1];
  const long long* ei    = (const long long*)d_in[2];   // int64 [2, E]
  const long long* batch = (const long long*)d_in[3];   // int64 [N]
  const float*     W1    = (const float*)d_in[4];
  const float*     b1    = (const float*)d_in[5];
  const float*     W2    = (const float*)d_in[6];
  const float*     b2    = (const float*)d_in[7];
  const float*     Wlin  = (const float*)d_in[8];
  const float*     blin  = (const float*)d_in[9];

  const int N = in_sizes[0];
  const int E = in_sizes[2] / 2;
  const long long* src = ei;
  const long long* dst = ei + E;

  // workspace carve-up
  char* base = (char*)d_ws;
  size_t off = 0;
  auto take = [&](size_t bytes) -> void* {
    void* p = base + off;
    off = (off + bytes + 255) & ~(size_t)255;
    return p;
  };
  const long NH = (long)N * HID;
  float*  deg  = (float*)take((size_t)N * 4);
  float*  dis  = (float*)take((size_t)N * 4);
  float*  h1   = (float*)take((size_t)NH * 4);
  float*  x1   = (float*)take((size_t)NH * 4);   // agg1 target -> x1 in place
  float*  h2   = (float*)take((size_t)NH * 4);
  float*  x2   = (float*)take((size_t)NH * 4);   // agg2 target -> x2 in place
  __bf16* w2p  = (__bf16*)take((size_t)HID * HID * 2);   // fragment-permuted W2
  float*  psum = (float*)take((size_t)(NGRAPH * NEXP + NGRAPH) * 4);
  float*  pcnt = psum + NGRAPH * NEXP;
  float*  out  = (float*)d_out;

  const int B = 256;
  auto cdiv = [](long a, long b) { return (int)((a + b - 1) / b); };

  // deterministic re-zero of atomic targets each call; psum/pcnt: 1 block zeroes only the first 256 of 576 floats
  k_zero<<<cdiv(N, B), B, 0, stream>>>(deg, N);
  k_zero<<<cdiv(NH, B), B, 0, stream>>>(x1, NH);
  k_zero<<<cdiv(NH, B), B, 0, stream>>>(x2, NH);
  k_zero<<<1, B, 0, stream>>>(psum, NGRAPH * NEXP + NGRAPH);

  k_permute_w<<<cdiv(HID * HID, B), B, 0, stream>>>(W2, w2p);
  k_deg<<<cdiv(E, B), B, 0, stream>>>(dst, deg, E);
  k_dis<<<cdiv(N, B), B, 0, stream>>>(deg, dis, N);

  // layer 1: tiny GEMM + scatter + self-loop/bias/ReLU epilogue
  k_fc1<<<cdiv(NH, B), B, 0, stream>>>(an, pos, W1, h1, N);
  k_agg<<<cdiv((long)E * 8, B), B, 0, stream>>>(src, dst, dis, h1, x1, E);
  k_self_bias<<<cdiv(NH, B), B, 0, stream>>>(h1, dis, b1, x1, N, 1);

  // layer 2: WMMA GEMM + scatter + self-loop/bias epilogue
  k_gemm_wmma<<<cdiv(N, 64), 256, 0, stream>>>(x1, w2p, h2, N);
  k_agg<<<cdiv((long)E * 8, B), B, 0, stream>>>(src, dst, dis, h2, x2, E);
  k_self_bias<<<cdiv(NH, B), B, 0, stream>>>(h2, dis, b2, x2, N, 0);

  // head: projection fused with pooling, then mean + log_softmax
  k_lin_pool<<<cdiv(N, B), B, 0, stream>>>(x2, Wlin, blin, batch, psum, pcnt, N);
  k_finalize<<<1, 64, 0, stream>>>(psum, pcnt, out);
}
